// LearnedIRConvolver_24661702213852
// MI455X (gfx1250) — matrix operand order repaired and checked by reading, not compiled, not run on hardware
//
#include <hip/hip_runtime.h>

typedef __attribute__((ext_vector_type(16))) _Float16     v16h;
typedef __attribute__((ext_vector_type(8)))  float        v8f;
typedef __attribute__((ext_vector_type(8)))  unsigned int v8u;

namespace {
constexpr int kIrLen   = 44100;
constexpr int kBatch   = 32;
constexpr int kT       = 441000;
constexpr int kNT      = 1408;            // 32-tap steps, multiple of 16 (branch-free)
constexpr int kChunks  = 88;              // kNT / 16 exactly
constexpr int kXPad    = 44104;           // front zero-pad of f16 x (mult of 8, >= L-1)
constexpr int kPL      = 486032;          // padded per-batch f16 x length (mult of 8)
constexpr int kIrvRuns = 1423;            // 32-tap runs in padded reversed IR (t+i <= 1422)
constexpr int kIrvU16  = kIrvRuns * 40;   // 80B per run: 64B data + 16B bank-stagger pad
constexpr int kCopyU16 = 1104;            // u16 per phase copy (>= 1088 staged + slack)
constexpr int kSpanPB  = 431;             // 1024-sample output spans per batch
constexpr int kJobs    = kBatch * kSpanPB; // 13792 wave jobs (4 WMMA tiles each)
constexpr int kWaves   = 16;
constexpr int kThreads = kWaves * 32;
constexpr int kWGs     = kJobs / kWaves;  // 862 exactly
} // namespace

// ---------------------------------------------------------------------------
// Prep: x (f32) -> zero-padded f16 stream per batch in workspace.
// ---------------------------------------------------------------------------
__global__ void prep_x_f16(const float* __restrict__ x, _Float16* __restrict__ xf) {
    int idx = blockIdx.x * blockDim.x + threadIdx.x;
    const int total = kBatch * kPL;
    if (idx >= total) return;
    int b = idx / kPL;
    int p = idx - b * kPL;
    int n = p - kXPad;
    float v = 0.0f;
    if ((unsigned)n < (unsigned)kT) v = x[b * kT + n];
    xf[idx] = (_Float16)v;
}

// ---------------------------------------------------------------------------
// Main: implicit-Toeplitz GEMM FIR via v_wmma_f32_16x16x32_f16.
// Each wave: one (batch, 1024-span) job = 4 WMMA tiles sharing the A operand,
// branch-free 16-step chunks, software-pipelined one step deep (ping-pong regs).
// Outputs of tile base m0: m = m0 + j - 32*i  (i = A-row, j = B/D-column).
// ---------------------------------------------------------------------------
__global__ __launch_bounds__(kThreads) void fir_wmma(
    const float* __restrict__ x, const float* __restrict__ ir,
    const float* __restrict__ wetp, const _Float16* __restrict__ xf,
    float* __restrict__ out)
{
    extern __shared__ __align__(16) char smem[];
    unsigned short* s_ir = (unsigned short*)smem;     // padded reversed IR (f16 bits)

    // ---- Cooperative build of reversed+padded IR in LDS (80B per 32-tap run).
    for (int u = threadIdx.x; u < kIrvRuns * 32; u += kThreads) {
        float v = 0.0f;
        if (u >= 480 && u < 480 + kIrLen) v = ir[44579 - u];  // irv[u] = ir[L-1-(u-480)]
        _Float16 h = (_Float16)v;
        s_ir[(u >> 5) * 40 + (u & 31)] = __builtin_bit_cast(unsigned short, h);
    }
    __syncthreads();

    const int lane = threadIdx.x & 31;
    const int wid  = threadIdx.x >> 5;
    const int j    = lane & 15;   // B/D column == A row index
    const int sg   = lane >> 4;   // K-half select (A and B), M-half select (D)

    const int P = blockIdx.x * kWaves + wid;       // grid is exact: P < kJobs
    const int b = P / kSpanPB;
    const int u = P - b * kSpanPB;
    const int M = u * 1024 + 480;                  // tile bases M+{0,16,512,528}

    const _Float16* xrowf = xf + b * kPL;

    // This wave's 4 phase copies: byte offset inside dynamic LDS.
    const unsigned cofs = (unsigned)(kIrvU16 * 2) + (unsigned)wid * (4 * kCopyU16 * 2);
    char* cbase = smem + cofs;

    // Per-lane B addressing: chunk-local u16 offset o = 5 + off_k + j + 16*sg + 32*tt.
    const int obase = 5 + j + 16 * sg;
    const int Q     = obase & 3;                   // phase copy (same for all 4 tiles)
    unsigned bo = cofs + (unsigned)(Q * (kCopyU16 * 2) + (obase - Q) * 2); // 8B aligned
    asm("" : "+v"(bo));   // pin base offset: force immediate-offset DS loads off it
    const char* bsel = smem + bo;

    // Per-lane A addressing: run (t + i)*80B, +32B per K-half, loads at +0/+16.
    const char* abase = (const char*)s_ir + j * 80 + sg * 32;

    v8f acc[4];
    acc[0] = v8f{}; acc[1] = v8f{}; acc[2] = v8f{}; acc[3] = v8f{};

    v16h a_op[2];
    v16h b_op[2][4];

    auto load_ops = [&](int t, int tt, int p) {
        const char* ap = abase + t * 80;
        const uint4 alo = *(const uint4*)(ap);
        const uint4 ahi = *(const uint4*)(ap + 16);
        v8u av;
        av[0] = alo.x; av[1] = alo.y; av[2] = alo.z; av[3] = alo.w;
        av[4] = ahi.x; av[5] = ahi.y; av[6] = ahi.z; av[7] = ahi.w;
        a_op[p] = __builtin_bit_cast(v16h, av);
        const char* bp = bsel + tt * 64;           // tt static under full unroll
        #pragma unroll
        for (int k = 0; k < 4; ++k) {
            const char* bk = bp + (k & 1) * 32 + (k >> 1) * 1024;
            const uint2 q0 = *(const uint2*)(bk +  0);
            const uint2 q1 = *(const uint2*)(bk +  8);
            const uint2 q2 = *(const uint2*)(bk + 16);
            const uint2 q3 = *(const uint2*)(bk + 24);
            v8u bv;
            bv[0] = q0.x; bv[1] = q0.y; bv[2] = q1.x; bv[3] = q1.y;
            bv[4] = q2.x; bv[5] = q2.y; bv[6] = q3.x; bv[7] = q3.y;
            b_op[p][k] = __builtin_bit_cast(v16h, bv);
        }
    };

    for (int c = 0; c < kChunks; ++c) {
        // ---- Stage 1088 f16 of x into 4 phase-shifted LDS copies (aligned I/O).
        const int G  = kXPad + M - 44579 + 512 * c;  // G mod 8 == 5 always
        const int Ga = G & ~7;
        const _Float16* src = xrowf + Ga;
        __builtin_prefetch(src + 1088 + 8 * lane, 0, 0);  // next-chunk warmup

        auto stage_group = [&](int g) {
            const uint4 q  = *(const uint4*)(src + 8 * g);
            const uint2 nx = *(const uint2*)(src + 8 * g + 8);
            uint4 c1, c2, c3;
            c2.x = q.y; c2.y = q.z; c2.z = q.w; c2.w = nx.x;
            c1.x = (q.x >> 16) | (q.y << 16);
            c1.y = (q.y >> 16) | (q.z << 16);
            c1.z = (q.z >> 16) | (q.w << 16);
            c1.w = (q.w >> 16) | (nx.x << 16);
            c3.x = (q.y >> 16) | (q.z << 16);
            c3.y = (q.z >> 16) | (q.w << 16);
            c3.z = (q.w >> 16) | (nx.x << 16);
            c3.w = (nx.x >> 16) | (nx.y << 16);
            *(uint4*)(cbase + 0 * (kCopyU16 * 2) + g * 16) = q;
            *(uint4*)(cbase + 1 * (kCopyU16 * 2) + g * 16) = c1;
            *(uint4*)(cbase + 2 * (kCopyU16 * 2) + g * 16) = c2;
            *(uint4*)(cbase + 3 * (kCopyU16 * 2) + g * 16) = c3;
        };
        #pragma unroll
        for (int r = 0; r < 4; ++r) stage_group(r * 32 + lane);  // 128 groups
        if (lane < 8) stage_group(128 + lane);                   // tail -> 136 groups
        // Same-wave LDS is in-order: stores above are seen by the loads below.

        const int tbase = c * 16;
        load_ops(tbase, 0, 0);                       // pipeline prologue
        #pragma unroll
        for (int tt = 0; tt < 16; ++tt) {            // fully unrolled, branch-free
            const int p = tt & 1;
            if (tt + 1 < 16) load_ops(tbase + tt + 1, tt + 1, p ^ 1);
            #pragma unroll
            for (int k = 0; k < 4; ++k)
                acc[k] = __builtin_amdgcn_wmma_f32_16x16x32_f16(
                    false, a_op[p], false, b_op[p][k], (short)0, acc[k],
                    false, false);
        }
    }

    // ---- Epilogue: dry/wet mix, masked scatter (D: lane j, rows r + 8*sg).
    const float wet = 1.0f / (1.0f + __expf(-wetp[0]));
    const float dry = 1.0f - wet;
    const float* xrow = x + b * kT;
    float* orow = out + b * kT;
    #pragma unroll
    for (int k = 0; k < 4; ++k) {
        const int off = (k & 1) * 16 + (k >> 1) * 512;
        #pragma unroll
        for (int r = 0; r < 8; ++r) {
            const int m = M + off + j - 32 * r - 256 * sg;
            if ((unsigned)m < (unsigned)kT)
                orow[m] = dry * xrow[m] + wet * acc[k][r];
        }
    }
}

// ---------------------------------------------------------------------------
extern "C" void kernel_launch(void* const* d_in, const int* in_sizes, int n_in,
                              void* d_out, int out_size, void* d_ws, size_t ws_size,
                              hipStream_t stream) {
    const float* x  = (const float*)d_in[0];   // (32,1,441000) f32
    const float* ir = (const float*)d_in[1];   // (44100,)      f32
    const float* wp = (const float*)d_in[2];   // (1,)          f32
    float* out      = (float*)d_out;
    _Float16* xf    = (_Float16*)d_ws;         // 32*kPL f16 (~29.7 MiB)

    const int total = kBatch * kPL;
    prep_x_f16<<<(total + 255) / 256, 256, 0, stream>>>(x, xf);

    const size_t shmem = (size_t)kIrvU16 * 2 + (size_t)kWaves * 4 * kCopyU16 * 2;
    fir_wmma<<<kWGs, kThreads, shmem, stream>>>(x, ir, wp, xf, out);
}
